// Net_27547920236745
// MI455X (gfx1250) — matrix operand order repaired and checked by reading, not compiled, not run on hardware
//
#include <hip/hip_runtime.h>
#include <hip/hip_bf16.h>
#include <math.h>

typedef __attribute__((ext_vector_type(16))) _Float16 v16h;
typedef __attribute__((ext_vector_type(8)))  _Float16 v8h;
typedef __attribute__((ext_vector_type(8)))  float    v8f;
typedef __attribute__((ext_vector_type(4)))  int      v4i;

#define T_STEPS 2048
#define BATCH   64
#define HID     256   // == input dim

#if __has_builtin(__builtin_amdgcn_global_load_async_to_lds_b128) && \
    __has_builtin(__builtin_amdgcn_s_wait_asynccnt)
#define USE_ASYNC_LDS 1
#else
#define USE_ASYNC_LDS 0
#endif

// ---------------------------------------------------------------------------
// Branchless tanh: native v_tanh_f32 on gfx1250 (confirmed lowering).
// ---------------------------------------------------------------------------
__device__ __forceinline__ float fast_tanh(float x) {
#if __has_builtin(__builtin_amdgcn_tanhf)
    return __builtin_amdgcn_tanhf(x);
#elif __has_builtin(__builtin_amdgcn_tanh_f32)
    return __builtin_amdgcn_tanh_f32(x);
#else
    float e = __builtin_amdgcn_exp2f(x * 2.885390081777927f);
    return 1.0f - 2.0f * __builtin_amdgcn_rcpf(1.0f + e);
#endif
}

// ---------------------------------------------------------------------------
// Fragment loaders (wave32 WMMA 16x16x32 f16 layouts, ISA 7.12.2)
// ---------------------------------------------------------------------------
__device__ __forceinline__ v16h load_a_frag(const _Float16* rowk /*&A[m][kb]*/, int half) {
    const v8h lo = *(const v8h*)(rowk + half * 16);
    const v8h hi = *(const v8h*)(rowk + half * 16 + 8);
    v16h a;
#pragma unroll
    for (int i = 0; i < 8; ++i) { a[i] = lo[i]; a[i + 8] = hi[i]; }
    return a;
}

__device__ __forceinline__ v16h load_b_frag(const _Float16* wrow /*&W[n][kb+half*16]*/) {
    const v8h lo = *(const v8h*)(wrow);
    const v8h hi = *(const v8h*)(wrow + 8);
    v16h b;
#pragma unroll
    for (int i = 0; i < 8; ++i) { b[i] = lo[i]; b[i + 8] = hi[i]; }
    return b;
}

__device__ __forceinline__ v8f wmma_f16(v16h a, v16h b, v8f c) {
    return __builtin_amdgcn_wmma_f32_16x16x32_f16(false, a, false, b, (short)0, c, false, false);
}

// ---------------------------------------------------------------------------
// Prep: weights f32 -> f16 in native [N][K] row-major (B-fragment layout),
// fuse bias pairs.
// ---------------------------------------------------------------------------
__global__ void prep_kernel(const float* Wih0, const float* Whh0,
                            const float* Wih1, const float* Whh1,
                            const float* bih0, const float* bhh0,
                            const float* bih1, const float* bhh1,
                            _Float16* Wih0h, _Float16* Whh0h,
                            _Float16* Wih1h, _Float16* Whh1h,
                            float* bias0, float* bias1) {
    int i = blockIdx.x * blockDim.x + threadIdx.x;
    if (i < HID * HID) {
        Wih0h[i] = (_Float16)Wih0[i];
        Whh0h[i] = (_Float16)Whh0[i];
        Wih1h[i] = (_Float16)Wih1[i];
        Whh1h[i] = (_Float16)Whh1[i];
    }
    if (i < HID) {
        bias0[i] = bih0[i] + bhh0[i];
        bias1[i] = bih1[i] + bhh1[i];
    }
}

// ---------------------------------------------------------------------------
// Wide input GEMM: out[r][n] = f16(sum_k A[r][k]*W[n][k] + bias[n])
// One block = 128 rows x 256 cols, 8 waves. LDS = 64KB A + 128KB W = 192KB.
// ---------------------------------------------------------------------------
__device__ __forceinline__ void stage_a_rows(const float* Ag, _Float16* As) {
    int t0 = threadIdx.x * 128;
    const float4* src = (const float4*)Ag;
#pragma unroll
    for (int v = 0; v < 32; ++v) {
        float4 f = src[(t0 >> 2) + v];
        _Float16* d = As + t0 + 4 * v;
        d[0] = (_Float16)f.x; d[1] = (_Float16)f.y;
        d[2] = (_Float16)f.z; d[3] = (_Float16)f.w;
    }
}
__device__ __forceinline__ void stage_a_rows(const _Float16* Ag, _Float16* As) {
    const uint4* src = (const uint4*)Ag;
    uint4* dst = (uint4*)As;
#pragma unroll
    for (int v = 0; v < 16; ++v) dst[threadIdx.x + 256 * v] = src[threadIdx.x + 256 * v];
}

template <typename TA>
__global__ void __launch_bounds__(256, 1)
gemm_bias_kernel(const TA* __restrict__ A,        // [T*B][256]
                 const _Float16* __restrict__ W,  // [256][256]
                 const float* __restrict__ bias,  // [256]
                 _Float16* __restrict__ out) {    // [T*B][256] f16
    extern __shared__ char smem[];
    _Float16* As = (_Float16*)smem;                   // 128*256
    _Float16* Ws = (_Float16*)(smem + 128 * 256 * 2); // 256*256

    const int rb = blockIdx.x;
    stage_a_rows(A + (size_t)rb * 128 * 256, As);
    {   // stage W: 8192 uint4, 32 per thread
        const uint4* src = (const uint4*)W;
        uint4* dst = (uint4*)Ws;
#pragma unroll
        for (int v = 0; v < 32; ++v) dst[threadIdx.x + 256 * v] = src[threadIdx.x + 256 * v];
    }
    __syncthreads();

    const int lane = threadIdx.x & 31;
    const int wv   = threadIdx.x >> 5;
    const int m16  = lane & 15;
    const int half = lane >> 4;

    const _Float16* Arow = As + (wv * 16 + m16) * 256;
    v16h afr[8];
#pragma unroll
    for (int kt = 0; kt < 8; ++kt) afr[kt] = load_a_frag(Arow + kt * 32, half);

#pragma unroll 4
    for (int nt = 0; nt < 16; ++nt) {
        const int n = nt * 16 + m16;
        const _Float16* Wrow = Ws + n * 256 + half * 16;
        v8f c = {};
#pragma unroll
        for (int kt = 0; kt < 8; ++kt) c = wmma_f16(afr[kt], load_b_frag(Wrow + kt * 32), c);

        const float bv = bias[n];
        const size_t gm = (size_t)(rb * 128 + wv * 16 + 8 * half) * 256 + n;
#pragma unroll
        for (int r = 0; r < 8; ++r) out[gm + (size_t)r * 256] = (_Float16)(c[r] + bv);
    }
}

// ---------------------------------------------------------------------------
// Async prefetch of one step's xp slice (64x256 f16 = 32KB) into LDS.
// 2048 b128 transfers -> 4 per thread (512 threads), ASYNCcnt-tracked.
// ---------------------------------------------------------------------------
__device__ __forceinline__ void prefetch_xp(const _Float16* src, _Float16* dstLds, int tid) {
#if USE_ASYNC_LDS
    const char* s = (const char*)src;
    char* d = (char*)dstLds;
#pragma unroll
    for (int v = 0; v < 4; ++v) {
        int off = (tid + 512 * v) * 16;
        __builtin_amdgcn_global_load_async_to_lds_b128(
            (v4i*)(s + off), (v4i*)(d + off), 0, 0);
    }
#else
    const uint4* s = (const uint4*)src;
    uint4* d = (uint4*)dstLds;
#pragma unroll
    for (int v = 0; v < 4; ++v) d[tid + 512 * v] = s[tid + 512 * v];
#endif
}

__device__ __forceinline__ void wait_async() {
#if USE_ASYNC_LDS
    __builtin_amdgcn_s_wait_asynccnt(0);
#endif
}

// ---------------------------------------------------------------------------
// Sequential recurrence: h_t = tanh(xp_t + h_{t-1} @ W_hh^T)
// One persistent workgroup, 512 threads = 16 wave32s, LDS-bound to 1 WG/WGP.
// LDS: W_hh 128KB + h double buffer 2x32KB + xp double buffer 2x32KB = 256KB.
// kt-outer / 4-N-tile-inner: one live A frag, 4 independent WMMA chains.
// ALL base pointers hoisted out of the t-loop; every ds access in the loop is
// base + small immediate (<=496B for W/A frags, <=3.7KB for xp/h epilogue).
// MODE 0: emit h sequence as f16; MODE 1: emit h sequence as f32.
// ---------------------------------------------------------------------------
template <int MODE>
__global__ void __launch_bounds__(512, 1)
recur_kernel(const _Float16* __restrict__ xp,    // [T*B][256] f16
             const _Float16* __restrict__ Whh,   // [256][256]
             _Float16* __restrict__ hseq16,      // MODE 0 output
             float* __restrict__ out32) {        // MODE 1 output
    extern __shared__ char smem[];
    _Float16* Ws  = (_Float16*)smem;          // 65536 halfs (128KB)
    _Float16* hA  = Ws + 256 * 256;           // 16384 halfs (32KB)
    _Float16* hB  = hA + 64 * 256;
    _Float16* xpA = hB + 64 * 256;            // 16384 halfs (32KB)
    _Float16* xpB = xpA + 64 * 256;

    const int tid = threadIdx.x;
    {   // stage W_hh (8192 uint4, 16/thread) and zero h_{-1}
        const uint4* src = (const uint4*)Whh;
        uint4* dst = (uint4*)Ws;
#pragma unroll
        for (int v = 0; v < 16; ++v) dst[tid + 512 * v] = src[tid + 512 * v];
        unsigned* z = (unsigned*)hA;
#pragma unroll
        for (int v = 0; v < 16; ++v) z[tid + 512 * v] = 0u;
    }
    prefetch_xp(xp, xpA, tid);     // step 0
    wait_async();
    __syncthreads();

    const int lane = tid & 31;
    const int wv   = tid >> 5;           // 0..15
    const int mt   = wv & 3;             // M tile (batch)
    const int ntg  = wv >> 2;            // N tile group
    const int m16  = lane & 15;
    const int half = lane >> 4;
    const int mb   = mt * 16;
    const int mloc = mb + 8 * half;      // first output row this lane owns
    const int nb   = ntg * 64 + m16;     // first output col; tile i at +i*16

    // ---- loop-invariant base pointers (no address math inside the t-loop) --
    // B fragments: per-tile row pointer; kt slice at +kt*32 halfs (<=496B imm)
    const _Float16* Wrow[4];
#pragma unroll
    for (int i = 0; i < 4; ++i) Wrow[i] = Ws + (nb + i * 16) * 256 + half * 16;
    // A fragments: even/odd h buffer row pointers
    const _Float16* ArowA = hA + (mb + m16) * 256;
    const _Float16* ArowB = hB + (mb + m16) * 256;
    // xp reads: tile i at +i*16, row r at +r*256 (<=3.7KB imm)
    const _Float16* xlA = xpA + mloc * 256 + nb;
    const _Float16* xlB = xpB + mloc * 256 + nb;
    // h stores go to the *other* buffer than the one read
    _Float16* nstA = hA + mloc * 256 + nb;   // used when nxt == hA (odd t)
    _Float16* nstB = hB + mloc * 256 + nb;   // used when nxt == hB (even t)
    // running global output pointers (advance by BATCH*HID per step)
    _Float16* g16 = (MODE == 0) ? hseq16 + mloc * 256 + nb : nullptr;
    float*    g32 = (MODE == 1) ? out32  + mloc * 256 + nb : nullptr;
    const _Float16* xpsrc = xp + (size_t)BATCH * HID;  // step-1 prefetch source
    _Float16* xnxtArr[2] = { xpB, xpA };               // written at step t: buf (t+1)&1

    for (int t = 0; t < T_STEPS; ++t) {
        const _Float16* Arow = (t & 1) ? ArowB : ArowA;
        const _Float16* xl   = (t & 1) ? xlB   : xlA;
        _Float16*       nst  = (t & 1) ? nstA  : nstB;

        if (t + 1 < T_STEPS) {  // overlap next step's xp fetch with this GEMM
            prefetch_xp(xpsrc, xnxtArr[t & 1], tid);
            xpsrc += BATCH * HID;
        }

        v8f acc[4];
#pragma unroll
        for (int i = 0; i < 4; ++i) acc[i] = (v8f){};

#pragma unroll
        for (int kt = 0; kt < 8; ++kt) {
            const v16h a = load_a_frag(Arow + kt * 32, half);
#pragma unroll
            for (int i = 0; i < 4; ++i)
                acc[i] = wmma_f16(a, load_b_frag(Wrow[i] + kt * 32), acc[i]);
        }

#pragma unroll
        for (int i = 0; i < 4; ++i) {
#pragma unroll
            for (int r = 0; r < 8; ++r) {
                const int off = i * 16 + r * 256;
                float v = fast_tanh(acc[i][r] + (float)xl[off]);
                nst[off] = (_Float16)v;
                if (MODE == 0) g16[off] = (_Float16)v;
                else           g32[off] = v;
            }
        }
        if (MODE == 0) g16 += BATCH * HID; else g32 += BATCH * HID;

        wait_async();
        __syncthreads();   // h_t and prefetched xp visible before step t+1
    }
}

// ---------------------------------------------------------------------------
// Host launch
// ---------------------------------------------------------------------------
extern "C" void kernel_launch(void* const* d_in, const int* in_sizes, int n_in,
                              void* d_out, int out_size, void* d_ws, size_t ws_size,
                              hipStream_t stream) {
    const float* x     = (const float*)d_in[0];
    const float* Wih0  = (const float*)d_in[1];
    const float* Whh0  = (const float*)d_in[2];
    const float* bih0  = (const float*)d_in[3];
    const float* bhh0  = (const float*)d_in[4];
    const float* Wih1  = (const float*)d_in[5];
    const float* Whh1  = (const float*)d_in[6];
    const float* bih1  = (const float*)d_in[7];
    const float* bhh1  = (const float*)d_in[8];

    const size_t NROWS = (size_t)T_STEPS * BATCH;            // 131072
    char* ws = (char*)d_ws;
    _Float16*  xpf    = (_Float16*)ws;                       // NROWS*256 f16 = 64MB
    _Float16*  h1f16  = (_Float16*)(ws + NROWS * 256 * 2);   // 64MB
    _Float16*  wbase  = (_Float16*)(ws + 2 * NROWS * 256 * 2);
    _Float16*  Wih0h  = wbase;
    _Float16*  Whh0h  = wbase + 1 * 65536;
    _Float16*  Wih1h  = wbase + 2 * 65536;
    _Float16*  Whh1h  = wbase + 3 * 65536;
    float*     bias0  = (float*)(wbase + 4 * 65536);
    float*     bias1  = bias0 + 256;

    const size_t LDS_GEMM  = (size_t)(128 * 256 + 256 * 256) * 2;    // 192KB
    const size_t LDS_RECUR = (size_t)(256 * 256 + 4 * 64 * 256) * 2; // 256KB

    prep_kernel<<<256, 256, 0, stream>>>(Wih0, Whh0, Wih1, Whh1,
                                         bih0, bhh0, bih1, bhh1,
                                         Wih0h, Whh0h, Wih1h, Whh1h, bias0, bias1);

    // Layer 0
    gemm_bias_kernel<float><<<NROWS / 128, 256, LDS_GEMM, stream>>>(x, Wih0h, bias0, xpf);
    recur_kernel<0><<<1, 512, LDS_RECUR, stream>>>(xpf, Whh0h, h1f16, nullptr);

    // Layer 1
    gemm_bias_kernel<_Float16><<<NROWS / 128, 256, LDS_GEMM, stream>>>(h1f16, Wih1h, bias1, xpf);
    recur_kernel<1><<<1, 512, LDS_RECUR, stream>>>(xpf, Whh1h, nullptr, (float*)d_out);
}
